// bi_model_lit_69054484185191
// MI455X (gfx1250) — matrix operand order repaired and checked by reading, not compiled, not run on hardware
//
#include <hip/hip_runtime.h>
#include <hip/hip_bf16.h>

// ---------------- types ----------------
typedef __attribute__((ext_vector_type(16))) _Float16 v16h;
typedef __attribute__((ext_vector_type(8)))  _Float16 v8h;
typedef __attribute__((ext_vector_type(8)))  float    v8f;

#define DIM   192
#define NH    6
#define HD    32
#define N3    576
#define HID   768
#define NTOK  131072   // B*H*W = 2*256*256
#define NWIN  2048     // 2 * 32 * 32
#define QSCALE 0.17677669529663687f   // 32^-0.5

// ---------------- WMMA helpers ----------------
// A operand: 16x32 f16, row-major source with stride ld (halves).
// lane l<16 : row mBase+l,     K [k0..k0+15], k0=kBase
// lane l>=16: row mBase+l-16,  k0 = kBase+16
__device__ __forceinline__ v16h load_a16(const _Float16* base, int ld, int mBase,
                                         int kBase, int lane) {
  int row = mBase + (lane & 15);
  int k0  = kBase + ((lane >> 4) << 4);
  const _Float16* p = base + (size_t)row * ld + k0;
  v8h lo = *(const v8h*)(p);
  v8h hi = *(const v8h*)(p + 8);
  v16h r;
#pragma unroll
  for (int i = 0; i < 8; ++i) { r[i] = lo[i]; r[i + 8] = hi[i]; }
  return r;
}

// B operand: 32x16 f16, source stored N-major [n][k] with stride ld (halves).
// lane l: column nBase+(l&15), 16 contiguous K at kBase + 16*(l>>4)
__device__ __forceinline__ v16h load_b16(const _Float16* base, int ld, int nBase,
                                         int kBase, int lane) {
  int col = nBase + (lane & 15);
  int k0  = kBase + ((lane >> 4) << 4);
  return *(const v16h*)(base + (size_t)col * ld + k0);
}

__device__ __forceinline__ v8f wmma_f16(v16h a, v16h b, v8f c) {
  return __builtin_amdgcn_wmma_f32_16x16x32_f16(false, a, false, b,
                                                (short)0, c, false, false);
}

__device__ __forceinline__ float gelu_exact(float v) {
  return 0.5f * v * (1.0f + erff(v * 0.70710678118654752f));
}

// ---------------- CDNA5 async global->LDS copy (ASYNCcnt path) ----------------
// LDS byte offset == low 32 bits of a generic pointer into shared memory
// (LDS aperture: addr[31:0] is the LDS address).
__device__ __forceinline__ void async_copy_b128(const void* gsrc, void* ldst) {
  unsigned int loff = (unsigned int)(unsigned long long)ldst;
  asm volatile("global_load_async_to_lds_b128 %0, %1, off"
               :: "v"(loff), "v"(gsrc) : "memory");
}
__device__ __forceinline__ void wait_async() {
#if __has_builtin(__builtin_amdgcn_s_wait_asynccnt)
  __builtin_amdgcn_s_wait_asynccnt(0);
#else
  asm volatile("s_wait_asynccnt 0" ::: "memory");
#endif
}

// ---------------- weight prep: transpose + convert to f16 ----------------
// src row-major [K][N] fp32 -> dst N-major [N][K] f16
__global__ void k_transpose_h(const float* __restrict__ src,
                              _Float16* __restrict__ dst, int K, int N) {
  int i = blockIdx.x * blockDim.x + threadIdx.x;
  if (i < K * N) {
    int n = i % N, k = i / N;
    dst[(size_t)n * K + k] = (_Float16)src[i];
  }
}

// ---------------- continuous relative position bias -> rpb[6][64][64] -------
__global__ void k_cpb(const float* __restrict__ scale,
                      const float* __restrict__ w1, const float* __restrict__ b1,
                      const float* __restrict__ w2, const float* __restrict__ b2,
                      float* __restrict__ rpb) {
  __shared__ float tab[225][NH];
  int t = threadIdx.x;
  float s0 = scale[0], s1 = scale[1];
  if (t < 225) {
    int a = t / 15, b = t % 15;
    float f0 = (float)(a - 7) * (8.0f / 7.0f);
    float f1 = (float)(b - 7) * (8.0f / 7.0f);
    float acc[NH] = {0, 0, 0, 0, 0, 0};
    for (int j = 0; j < 512; ++j) {
      float h = f0 * w1[j] + f1 * w1[512 + j] + s0 * w1[1024 + j] +
                s1 * w1[1536 + j] + b1[j];
      h = fmaxf(h, 0.0f);
#pragma unroll
      for (int k = 0; k < NH; ++k) acc[k] += h * w2[j * NH + k];
    }
#pragma unroll
    for (int k = 0; k < NH; ++k) tab[t][k] = acc[k] + b2[k];
  }
  __syncthreads();
  for (int e = t; e < NH * 64 * 64; e += blockDim.x) {
    int hh = e >> 12; int rem = e & 4095; int m = rem >> 6; int n = rem & 63;
    int idx = ((m >> 3) - (n >> 3) + 7) * 15 + ((m & 7) - (n & 7) + 7);
    rpb[e] = tab[idx][hh];
  }
}

// ---------------- LN1 + QKV GEMM (64 rows/block, 8 waves) ----------------
// writes q (scaled), k window-major row-major f16; v transposed per window.
__global__ void k_qkv(const float* __restrict__ x, const float* __restrict__ g,
                      const float* __restrict__ bln,
                      const _Float16* __restrict__ wt,
                      const float* __restrict__ bias,
                      _Float16* __restrict__ qw, _Float16* __restrict__ kw,
                      _Float16* __restrict__ vtw) {
  __shared__ __align__(32) _Float16 xln[64][DIM];
  int tid = threadIdx.x, lane = tid & 31, wv = tid >> 5;
  int r0 = blockIdx.x * 64;
  if (tid < 64) {
    const float* xr = x + (size_t)(r0 + tid) * DIM;
    float sum = 0.0f, sq = 0.0f;
    for (int c = 0; c < DIM; ++c) { float v = xr[c]; sum += v; sq += v * v; }
    float mean = sum * (1.0f / DIM);
    float var  = sq * (1.0f / DIM) - mean * mean;
    float inv  = rsqrtf(var + 1e-5f);
    for (int c = 0; c < DIM; ++c)
      xln[tid][c] = (_Float16)(((xr[c] - mean) * inv) * g[c] + bln[c]);
  }
  __syncthreads();
  // 4 m-tiles x 36 n-tiles = 144 tiles; 18 per wave
  for (int ti = 0; ti < 18; ++ti) {
    int tile = wv * 18 + ti;
    int mt = tile / 36, nt = tile % 36;
    v8f acc = {};
#pragma unroll
    for (int kk = 0; kk < 6; ++kk) {
      v16h a = load_a16(&xln[0][0], DIM, mt * 16, kk * 32, lane);
      v16h b = load_b16(wt, DIM, nt * 16, kk * 32, lane);
      acc = wmma_f16(a, b, acc);
    }
    int n = nt * 16 + (lane & 15);
    float bv = bias[n];
    float qs = (n < DIM) ? QSCALE : 1.0f;
#pragma unroll
    for (int r = 0; r < 8; ++r) {
      int m = mt * 16 + ((lane >> 4) << 3) + r;
      int t = r0 + m;
      int bi = t >> 16; int rem = t & 65535;
      int y = rem >> 8, xx = rem & 255;
      int win = bi * 1024 + (y >> 3) * 32 + (xx >> 3);
      int pos = ((y & 7) << 3) + (xx & 7);
      float val = (acc[r] + bv) * qs;
      if (n < DIM) {
        qw[(size_t)(win * 64 + pos) * DIM + n] = (_Float16)val;
      } else if (n < 2 * DIM) {
        kw[(size_t)(win * 64 + pos) * DIM + (n - DIM)] = (_Float16)val;
      } else {
        vtw[(size_t)win * (DIM * 64) + (size_t)(n - 2 * DIM) * 64 + pos] =
            (_Float16)val;
      }
    }
  }
}

// ---------------- windowed attention + proj + residual (1 block / window) ----
__global__ void k_attn(const _Float16* __restrict__ qw,
                       const _Float16* __restrict__ kw,
                       const _Float16* __restrict__ vtw,
                       const float* __restrict__ rpb,
                       const _Float16* __restrict__ pwt,
                       const float* __restrict__ pb,
                       const float* __restrict__ x, float* __restrict__ x1) {
  __shared__ __align__(32) float     s[64][64];
  __shared__ __align__(32) _Float16  sf[64][64];
  __shared__ __align__(32) _Float16  o[64][DIM];
  int tid = threadIdx.x, lane = tid & 31, wv = tid >> 5;   // 4 waves
  int win = blockIdx.x;
  const _Float16* qwin = qw + (size_t)win * 64 * DIM;
  const _Float16* kwin = kw + (size_t)win * 64 * DIM;
  const _Float16* vwin = vtw + (size_t)win * (DIM * 64);

  for (int h = 0; h < NH; ++h) {
    // scores = (q*s) @ k^T + rpb : 16 tiles, 4 per wave, K = HD = 32
    for (int q = 0; q < 4; ++q) {
      int tile = wv * 4 + q, mt = tile >> 2, nt = tile & 3;
      v16h a = load_a16(qwin, DIM, mt * 16, h * HD, lane);
      v16h b = load_b16(kwin, DIM, nt * 16, h * HD, lane);
      v8f acc = {};
      acc = wmma_f16(a, b, acc);
      int n = nt * 16 + (lane & 15);
#pragma unroll
      for (int r = 0; r < 8; ++r) {
        int m = mt * 16 + ((lane >> 4) << 3) + r;
        s[m][n] = acc[r] + rpb[h * 4096 + m * 64 + n];
      }
    }
    __syncthreads();
    // softmax per row
    if (tid < 64) {
      float row[64];
      float mx = -1e30f;
#pragma unroll
      for (int j = 0; j < 64; ++j) { row[j] = s[tid][j]; mx = fmaxf(mx, row[j]); }
      float sum = 0.0f;
#pragma unroll
      for (int j = 0; j < 64; ++j) { float e = __expf(row[j] - mx); row[j] = e; sum += e; }
      float inv = 1.0f / sum;
#pragma unroll
      for (int j = 0; j < 64; ++j) sf[tid][j] = (_Float16)(row[j] * inv);
    }
    __syncthreads();
    // out_h = attn @ v : 4x2 tiles, 2 per wave, K = 64
    for (int q = 0; q < 2; ++q) {
      int tile = wv * 2 + q, mt = tile >> 1, dt = tile & 1;
      v8f acc = {};
#pragma unroll
      for (int kk = 0; kk < 2; ++kk) {
        v16h a = load_a16(&sf[0][0], 64, mt * 16, kk * 32, lane);
        v16h b = load_b16(vwin, 64, h * HD + dt * 16, kk * 32, lane);
        acc = wmma_f16(a, b, acc);
      }
      int d = dt * 16 + (lane & 15);
#pragma unroll
      for (int r = 0; r < 8; ++r) {
        int m = mt * 16 + ((lane >> 4) << 3) + r;
        o[m][h * HD + d] = (_Float16)acc[r];
      }
    }
    __syncthreads();
  }
  // proj (64x192x192) + bias + shortcut, window-reverse store
  int bi = win >> 10, wr = (win >> 5) & 31, wc = win & 31;
  for (int q = 0; q < 12; ++q) {
    int tile = wv * 12 + q, mt = tile / 12, nt = tile % 12;
    v8f acc = {};
#pragma unroll
    for (int kk = 0; kk < 6; ++kk) {
      v16h a = load_a16(&o[0][0], DIM, mt * 16, kk * 32, lane);
      v16h b = load_b16(pwt, DIM, nt * 16, kk * 32, lane);
      acc = wmma_f16(a, b, acc);
    }
    int n = nt * 16 + (lane & 15);
    float bv = pb[n];
#pragma unroll
    for (int r = 0; r < 8; ++r) {
      int m = mt * 16 + ((lane >> 4) << 3) + r;
      int y = wr * 8 + (m >> 3), xx = wc * 8 + (m & 7);
      size_t t = (size_t)bi * 65536 + (size_t)y * 256 + xx;
      x1[t * DIM + n] = acc[r] + bv + x[t * DIM + n];
    }
  }
}

// ---------------- LN2 + fc1 + GELU (f16 out) ----------------
__global__ void k_fc1(const float* __restrict__ x1, const float* __restrict__ g,
                      const float* __restrict__ bln,
                      const _Float16* __restrict__ wt,
                      const float* __restrict__ bias, _Float16* __restrict__ h1) {
  __shared__ __align__(32) _Float16 xln[64][DIM];
  int tid = threadIdx.x, lane = tid & 31, wv = tid >> 5;
  int r0 = blockIdx.x * 64;
  if (tid < 64) {
    const float* xr = x1 + (size_t)(r0 + tid) * DIM;
    float sum = 0.0f, sq = 0.0f;
    for (int c = 0; c < DIM; ++c) { float v = xr[c]; sum += v; sq += v * v; }
    float mean = sum * (1.0f / DIM);
    float var  = sq * (1.0f / DIM) - mean * mean;
    float inv  = rsqrtf(var + 1e-5f);
    for (int c = 0; c < DIM; ++c)
      xln[tid][c] = (_Float16)(((xr[c] - mean) * inv) * g[c] + bln[c]);
  }
  __syncthreads();
  // 4 m-tiles x 48 n-tiles = 192 tiles; 24 per wave
  for (int ti = 0; ti < 24; ++ti) {
    int tile = wv * 24 + ti;
    int mt = tile / 48, nt = tile % 48;
    v8f acc = {};
#pragma unroll
    for (int kk = 0; kk < 6; ++kk) {
      v16h a = load_a16(&xln[0][0], DIM, mt * 16, kk * 32, lane);
      v16h b = load_b16(wt, DIM, nt * 16, kk * 32, lane);
      acc = wmma_f16(a, b, acc);
    }
    int n = nt * 16 + (lane & 15);
    float bv = bias[n];
#pragma unroll
    for (int r = 0; r < 8; ++r) {
      int m = mt * 16 + ((lane >> 4) << 3) + r;
      size_t t = (size_t)(r0 + m);
      h1[t * HID + n] = (_Float16)gelu_exact(acc[r] + bv);
    }
  }
}

// ---------------- depthwise 3x3 + bias + GELU (LDS row staging, async) -------
// one block = one image row (256 px) x 8 channels
__global__ void k_dwconv(const _Float16* __restrict__ h1,
                         const float* __restrict__ w,
                         const float* __restrict__ bias,
                         _Float16* __restrict__ h2) {
  __shared__ __align__(16) _Float16 ld[3][256][8];   // rows y-1,y,y+1
  __shared__ float wk[8][9];
  __shared__ float bk[8];
  int tid = threadIdx.x;
  int c0 = blockIdx.y * 8;
  if (tid < 72) { int c = tid / 9, k = tid % 9; wk[c][k] = w[(c0 + c) * 9 + k]; }
  if (tid < 8) bk[tid] = bias[c0 + tid];
  int p = blockIdx.x * 256 + tid;            // pixel over B*H*W
  int b = p >> 16, y = (p >> 8) & 255, xx = p & 255;
#pragma unroll
  for (int r = 0; r < 3; ++r) {
    int yy = y + r - 1;                      // uniform across block
    if (yy >= 0 && yy <= 255) {
      const _Float16* src =
          h1 + ((size_t)b * 65536 + (size_t)yy * 256 + xx) * HID + c0;
      async_copy_b128(src, &ld[r][xx][0]);
    } else {
      v8h z = {};
      *(v8h*)&ld[r][xx][0] = z;
    }
  }
  wait_async();
  __syncthreads();
  float acc[8];
#pragma unroll
  for (int c = 0; c < 8; ++c) acc[c] = bk[c];
#pragma unroll
  for (int ky = 0; ky < 3; ++ky) {
#pragma unroll
    for (int kx = 0; kx < 3; ++kx) {
      int xk = xx + kx - 1;
      if (xk < 0 || xk > 255) continue;
      v8h vs = *(const v8h*)&ld[ky][xk][0];
#pragma unroll
      for (int c = 0; c < 8; ++c) acc[c] += wk[c][ky * 3 + kx] * (float)vs[c];
    }
  }
  _Float16* dst = h2 + (size_t)p * HID + c0;
#pragma unroll
  for (int c = 0; c < 8; ++c) dst[c] = (_Float16)gelu_exact(acc[c]);
}

// ---------------- fc2 + bias + residual (K-chunked, async A staging) --------
__global__ void k_fc2(const _Float16* __restrict__ h2,
                      const _Float16* __restrict__ wt,
                      const float* __restrict__ bias,
                      const float* __restrict__ x1, float* __restrict__ out) {
  __shared__ __align__(32) _Float16 a_s[64][DIM];    // 64 x 192 K-chunk of A
  int tid = threadIdx.x, lane = tid & 31, wv = tid >> 5;
  int r0 = blockIdx.x * 64;
  v8f acc[6];
#pragma unroll
  for (int i = 0; i < 6; ++i) { v8f z = {}; acc[i] = z; }
  // K = 768 in 4 chunks of 192; A staged once per chunk -> h2 read exactly once
  for (int kc = 0; kc < 4; ++kc) {
    __syncthreads();                         // protect a_s reuse
#pragma unroll
    for (int i = 0; i < 6; ++i) {            // 1536 b128 copies / 256 threads
      int idx = tid + i * 256;
      int row = idx / 24, grp = idx % 24;
      const _Float16* src = h2 + (size_t)(r0 + row) * HID + kc * DIM + grp * 8;
      async_copy_b128(src, &a_s[row][grp * 8]);
    }
    wait_async();
    __syncthreads();
    // 4 m-tiles x 12 n-tiles = 48 tiles; 6 per wave, accumulated across chunks
#pragma unroll
    for (int ti = 0; ti < 6; ++ti) {
      int tile = wv * 6 + ti;
      int mt = tile / 12, nt = tile % 12;
#pragma unroll
      for (int kk = 0; kk < 6; ++kk) {
        v16h a = load_a16(&a_s[0][0], DIM, mt * 16, kk * 32, lane);
        v16h b = load_b16(wt, HID, nt * 16, kc * DIM + kk * 32, lane);
        acc[ti] = wmma_f16(a, b, acc[ti]);
      }
    }
  }
#pragma unroll
  for (int ti = 0; ti < 6; ++ti) {
    int tile = wv * 6 + ti;
    int mt = tile / 12, nt = tile % 12;
    int n = nt * 16 + (lane & 15);
    float bv = bias[n];
#pragma unroll
    for (int r = 0; r < 8; ++r) {
      int m = mt * 16 + ((lane >> 4) << 3) + r;
      size_t t = (size_t)(r0 + m);
      out[t * DIM + n] = acc[ti][r] + bv + x1[t * DIM + n];
    }
  }
}

// ---------------- launch ----------------
extern "C" void kernel_launch(void* const* d_in, const int* in_sizes, int n_in,
                              void* d_out, int out_size, void* d_ws,
                              size_t ws_size, hipStream_t stream) {
  (void)in_sizes; (void)n_in; (void)out_size; (void)ws_size;
  const float* x      = (const float*)d_in[0];
  const float* scale  = (const float*)d_in[1];
  const float* n1g    = (const float*)d_in[2];
  const float* n1b    = (const float*)d_in[3];
  const float* qkv_w  = (const float*)d_in[4];
  const float* qkv_b  = (const float*)d_in[5];
  const float* proj_w = (const float*)d_in[6];
  const float* proj_b = (const float*)d_in[7];
  const float* cpb_w1 = (const float*)d_in[8];
  const float* cpb_b1 = (const float*)d_in[9];
  const float* cpb_w2 = (const float*)d_in[10];
  const float* cpb_b2 = (const float*)d_in[11];
  const float* n2g    = (const float*)d_in[12];
  const float* n2b    = (const float*)d_in[13];
  const float* fc1_w  = (const float*)d_in[14];
  const float* fc1_b  = (const float*)d_in[15];
  const float* fc2_w  = (const float*)d_in[16];
  const float* fc2_b  = (const float*)d_in[17];
  const float* dw_w   = (const float*)d_in[18];
  const float* dw_b   = (const float*)d_in[19];

  // workspace carve-up (all sizes multiples of 256 bytes)
  char* ws = (char*)d_ws;
  _Float16* qw      = (_Float16*)ws;  ws += (size_t)NTOK * DIM * 2;   // 50.3 MB
  _Float16* kw      = (_Float16*)ws;  ws += (size_t)NTOK * DIM * 2;   // 50.3 MB
  _Float16* vtw     = (_Float16*)ws;  ws += (size_t)NTOK * DIM * 2;   // 50.3 MB
  float*    x1      = (float*)ws;     ws += (size_t)NTOK * DIM * 4;   // 100.7 MB
  _Float16* h1      = (_Float16*)ws;  ws += (size_t)NTOK * HID * 2;   // 201.3 MB
  _Float16* h2      = (_Float16*)ws;  ws += (size_t)NTOK * HID * 2;   // 201.3 MB
  _Float16* qkv_wt  = (_Float16*)ws;  ws += (size_t)DIM * N3 * 2;
  _Float16* proj_wt = (_Float16*)ws;  ws += (size_t)DIM * DIM * 2;
  _Float16* fc1_wt  = (_Float16*)ws;  ws += (size_t)DIM * HID * 2;
  _Float16* fc2_wt  = (_Float16*)ws;  ws += (size_t)HID * DIM * 2;
  float*    rpb     = (float*)ws;     ws += (size_t)NH * 64 * 64 * 4;

  // weight prep (L2-resident after this)
  k_transpose_h<<<(DIM * N3 + 255) / 256, 256, 0, stream>>>(qkv_w, qkv_wt, DIM, N3);
  k_transpose_h<<<(DIM * DIM + 255) / 256, 256, 0, stream>>>(proj_w, proj_wt, DIM, DIM);
  k_transpose_h<<<(DIM * HID + 255) / 256, 256, 0, stream>>>(fc1_w, fc1_wt, DIM, HID);
  k_transpose_h<<<(HID * DIM + 255) / 256, 256, 0, stream>>>(fc2_w, fc2_wt, HID, DIM);
  k_cpb<<<1, 256, 0, stream>>>(scale, cpb_w1, cpb_b1, cpb_w2, cpb_b2, rpb);

  // attention branch
  k_qkv<<<NTOK / 64, 256, 0, stream>>>(x, n1g, n1b, qkv_wt, qkv_b, qw, kw, vtw);
  k_attn<<<NWIN, 128, 0, stream>>>(qw, kw, vtw, rpb, proj_wt, proj_b, x, x1);

  // MLP branch
  k_fc1<<<NTOK / 64, 256, 0, stream>>>(x1, n2g, n2b, fc1_wt, fc1_b, h1);
  dim3 gdw(NTOK / 256, HID / 8);
  k_dwconv<<<gdw, 256, 0, stream>>>(h1, dw_w, dw_b, h2);
  k_fc2<<<NTOK / 64, 256, 0, stream>>>(h2, fc2_wt, fc2_b, x1, (float*)d_out);
}
